// AttnDecoder_4999341932806
// MI455X (gfx1250) — matrix operand order repaired and checked by reading, not compiled, not run on hardware
//
#include <hip/hip_runtime.h>

typedef __attribute__((ext_vector_type(16))) __bf16 v16bf;
typedef __attribute__((ext_vector_type(8)))  float  v8f;

// POD vectors (HIP's uint4 has ctors -> not union-safe)
struct U4 { unsigned a, b, c, d; };
struct U2 { unsigned a, b; };
union UXbf8 { U4 q; __bf16 h[8]; };
union UXbf4 { U2 q; __bf16 h[4]; };

#define Bv 128
#define Tv 256
#define Ev 256
#define Dv 256

// Hardware transcendental tanh if the toolchain exposes it for gfx1250.
__device__ inline float fast_tanh(float x) {
#if __has_builtin(__builtin_amdgcn_tanhf)
  return __builtin_amdgcn_tanhf(x);
#elif __has_builtin(__builtin_amdgcn_tanh_f32)
  return __builtin_amdgcn_tanh_f32(x);
#else
  return tanhf(x);
#endif
}

__device__ inline float fast_sigmoid(float x) {
  return 1.f / (1.f + __expf(-x));
}

// ---------------------------------------------------------------- prep ----
__global__ void k_f32_to_bf16(const float* __restrict__ src, __bf16* __restrict__ dst, int n) {
  int i = blockIdx.x * blockDim.x + threadIdx.x;
  int stride = gridDim.x * blockDim.x;
  for (; i < n; i += stride) dst[i] = (__bf16)src[i];
}

// Pack W into WMMA B-fragment order for 16x16x32 bf16:
// logical B[k][n] = src[n*rowStride + colOff + k]  (i.e. B = W^T slice)
// dst layout: [(kt*(N/16)+nt)*512 + lane*16 + s], s=2v+e; K = 8*(lane>=16) + s for
// s<8, 16 + 8*(lane>=16) + (s-8) for s>=8; n = nt*16 + (lane&15).
__global__ void k_pack_b(const float* __restrict__ src, __bf16* __restrict__ dst,
                         int K, int N, int rowStride, int colOff) {
  int total = K * N;
  int nT = N >> 4;
  for (int i = blockIdx.x * blockDim.x + threadIdx.x; i < total; i += gridDim.x * blockDim.x) {
    int j = i & 511, blk = i >> 9;
    int nt = blk % nT, kt = blk / nT;
    int lane = j >> 4, e16 = j & 15;
    int v = e16 >> 1, eo = e16 & 1;
    int k = kt * 32 + ((lane & 16) ? 8 : 0) + ((e16 < 8) ? e16 : 8 + e16);
    int n = nt * 16 + (lane & 15);
    dst[i] = (__bf16)src[n * rowStride + colOff + k];
  }
}

// ----------------------------------------------------- enc_proj (+b1) -----
// enc_proj[b,t,:] = X[b,t,:] @ W1_enc^T + b1, stored bf16.
// M = 16 rows of (b*T+t), N = 256, K = 256. 8 waves x 2 n-tiles each.
__global__ __launch_bounds__(256)
void k_encproj(const __bf16* __restrict__ Xbf, const __bf16* __restrict__ Bpack,
               const float* __restrict__ b1, __bf16* __restrict__ encb) {
  int lane = threadIdx.x & 31;
  int wave = threadIdx.x >> 5;          // 0..7
  int rowBase = blockIdx.x << 4;        // of 32768
  int r = lane & 15;
  int half8 = (lane & 16) ? 8 : 0;
  const unsigned* X32 = (const unsigned*)Xbf;
  int nt0 = wave * 2;

  v8f acc0 = {}, acc1 = {};
#pragma unroll
  for (int kt = 0; kt < 8; ++kt) {
    union { v16bf v; unsigned u[8]; } a;
    int kb = kt * 32 + half8;
#pragma unroll
    for (int vv = 0; vv < 8; ++vv) {
      int k = kb + ((vv < 4) ? 2 * vv : 16 + 2 * (vv - 4));
      a.u[vv] = X32[((rowBase + r) * Ev + k) >> 1];
    }
    union { v16bf v; U4 q[2]; } bm0, bm1;
    const U4* bp0 = (const U4*)(Bpack + ((kt * 16 + nt0) * 512 + lane * 16));
    const U4* bp1 = (const U4*)(Bpack + ((kt * 16 + nt0 + 1) * 512 + lane * 16));
    bm0.q[0] = bp0[0]; bm0.q[1] = bp0[1];
    bm1.q[0] = bp1[0]; bm1.q[1] = bp1[1];
    acc0 = __builtin_amdgcn_wmma_f32_16x16x32_bf16(false, a.v, false, bm0.v, (short)0, acc0, false, false);
    acc1 = __builtin_amdgcn_wmma_f32_16x16x32_bf16(false, a.v, false, bm1.v, (short)0, acc1, false, false);
  }
#pragma unroll 1
  for (int p = 0; p < 2; ++p) {
    int col = (nt0 + p) * 16 + (lane & 15);
    float bb = b1[col];
    const v8f& acc = p ? acc1 : acc0;
#pragma unroll
    for (int vv = 0; vv < 8; ++vv) {
      int m = half8 + vv;
      encb[(size_t)(rowBase + m) * Ev + col] = (__bf16)(acc[vv] + bb);
    }
  }
}

// ------------------------------------------------------- persistent scan --
// 8 blocks x 1024 threads (32 waves). Each block owns 16 batch rows for all
// 256 timesteps; h/c/context live in LDS. WMMA phases use the waves they
// need; the tanh-bound score phase and pointwise phases use all 32 waves.
__global__ __launch_bounds__(1024)
void k_scan(const __bf16* __restrict__ Xbf, const __bf16* __restrict__ encb,
            const __bf16* __restrict__ packW1hc, const __bf16* __restrict__ packWhh,
            const float* __restrict__ y_history,
            const float* __restrict__ h0, const float* __restrict__ c0,
            const float* __restrict__ attn_w2, const float* __restrict__ attn_b2,
            const float* __restrict__ W_ih, const float* __restrict__ b_ih,
            const float* __restrict__ b_hh,
            const float* __restrict__ fc_W, const float* __restrict__ fc_b,
            const float* __restrict__ fcout_W, const float* __restrict__ fcout_b,
            float* __restrict__ out) {
  __shared__ __align__(16) float smem[38272];           // ~149.5 KB
  float* sPreCtx = smem;                                 // 4096 (pre, then ctx)
  float* sScore  = smem + 4096;                          // 4096
  float* sGates  = smem + 8192;                          // 16384
  float* sH      = smem + 24576;                         // 4096
  float* sC      = smem + 28672;                         // 4096
  __bf16* sHbf   = (__bf16*)(smem + 32768);              // 4096 bf16
  __bf16* sCbf   = (__bf16*)(smem + 34816);              // 4096 bf16
  float* sYt     = smem + 36864;                         // 128
  float* sW2     = smem + 36992;                         // 256
  float* sBias   = smem + 37248;                         // 1024
  const unsigned* sHbf32 = (const unsigned*)sHbf;
  const unsigned* sCbf32 = (const unsigned*)sCbf;

  int tid = threadIdx.x;
  int lane = tid & 31, wave = tid >> 5;   // 32 waves
  int b0 = blockIdx.x << 4;
  float ab2 = attn_b2[0];

  for (int i = tid; i < 4096; i += 1024) {
    int r = i >> 8, d = i & 255;
    float hv = h0[(b0 + r) * Dv + d];
    float cv = c0[(b0 + r) * Dv + d];
    sH[i] = hv; sC[i] = cv;
    sHbf[i] = (__bf16)hv; sCbf[i] = (__bf16)cv;
  }
  if (tid < 256) sW2[tid] = attn_w2[tid];
  if (tid < 1024) sBias[tid] = b_ih[tid] + b_hh[tid];
  __syncthreads();

  for (int t = 0; t < Tv; ++t) {
    // --- Phase A: pre = [h c] @ W1_hc^T  (M=16, N=256, K=512); waves 0..15
    if (wave < 16) {
      int nt = wave;
      int r = lane & 15;
      int half8 = (lane & 16) ? 8 : 0;
      v8f acc = {};
#pragma unroll
      for (int kt = 0; kt < 16; ++kt) {
        union { v16bf v; unsigned u[8]; } a;
        const unsigned* src = (kt < 8) ? sHbf32 : sCbf32;
        int kb = (kt & 7) * 32 + half8;
#pragma unroll
        for (int vv = 0; vv < 8; ++vv) {
          int k = kb + ((vv < 4) ? 2 * vv : 16 + 2 * (vv - 4));
          a.u[vv] = src[(r * 256 + k) >> 1];
        }
        union { v16bf v; U4 q[2]; } bm;
        const U4* bp = (const U4*)(packW1hc + ((kt * 16 + nt) * 512 + lane * 16));
        bm.q[0] = bp[0]; bm.q[1] = bp[1];
        acc = __builtin_amdgcn_wmma_f32_16x16x32_bf16(false, a.v, false, bm.v, (short)0, acc, false, false);
      }
      int col = nt * 16 + (lane & 15);
#pragma unroll
      for (int vv = 0; vv < 8; ++vv) sPreCtx[(half8 + vv) * 256 + col] = acc[vv];
    }
    __syncthreads();

    // --- Phase B: scores[r][tt] = b2 + sum_f w2[f]*tanh(pre[r][f]+encb[r,tt,f])
    //     all 32 waves; 4 (r,tt) pairs per thread; dominant phase.
#pragma unroll 1
    for (int i = 0; i < 4; ++i) {
      int idx = tid + i * 1024;
      int r = idx & 15, tt = idx >> 4;
      const __bf16* ep = encb + ((size_t)(b0 + r) * Tv + tt) * Ev;
      const float* pp = sPreCtx + r * 256;
      float s = ab2;
      for (int f0 = 0; f0 < 256; f0 += 8) {
        UXbf8 u; u.q = *(const U4*)(ep + f0);
#pragma unroll
        for (int j = 0; j < 8; ++j)
          s += sW2[f0 + j] * fast_tanh(pp[f0 + j] + (float)u.h[j]);
      }
      sScore[r * 256 + tt] = s;
    }
    __syncthreads();

    // --- Phase C: softmax over T per row; wave w (<16) handles row w
    if (wave < 16) {
      int r = wave;
      float m = -1e30f;
#pragma unroll
      for (int j = 0; j < 8; ++j) m = fmaxf(m, sScore[r * 256 + lane + j * 32]);
#pragma unroll
      for (int d = 16; d >= 1; d >>= 1) m = fmaxf(m, __shfl_xor(m, d, 32));
      float e[8]; float sum = 0.f;
#pragma unroll
      for (int j = 0; j < 8; ++j) { e[j] = __expf(sScore[r * 256 + lane + j * 32] - m); sum += e[j]; }
#pragma unroll
      for (int d = 16; d >= 1; d >>= 1) sum += __shfl_xor(sum, d, 32);
      float inv = 1.f / sum;
#pragma unroll
      for (int j = 0; j < 8; ++j) sScore[r * 256 + lane + j * 32] = e[j] * inv;
    }
    __syncthreads();

    // --- Phase D: context[r][e] = sum_t w[r][t] * X[r,t,e]; all waves
    {
      int r = tid >> 6;             // 16 rows x 64 threads
      int e0 = (tid & 63) * 4;      // 64 threads * 4 cols = 256
      float acc[4] = {};
      const __bf16* xb = Xbf + ((size_t)(b0 + r) * Tv) * Ev + e0;
      const float* wr = sScore + r * 256;
      for (int tt = 0; tt < 256; ++tt) {
        float w = wr[tt];
        UXbf4 u; u.q = *(const U2*)(xb + (size_t)tt * Ev);
#pragma unroll
        for (int j = 0; j < 4; ++j) acc[j] += w * (float)u.h[j];
      }
#pragma unroll
      for (int j = 0; j < 4; ++j) sPreCtx[r * 256 + e0 + j] = acc[j];
    }
    __syncthreads();

    // --- Phase E: y_tilde = [ctx, y_t] @ fc_W^T + fc_b  (16x8)
    if (tid < 128) {
      int r = tid >> 3, o = tid & 7;
      const float* w = fc_W + o * (Ev + 8);
      float acc = fc_b[o];
      const float* cx = sPreCtx + r * 256;
      for (int e = 0; e < 256; ++e) acc += cx[e] * w[e];
      const float* yt = y_history + ((size_t)(b0 + r) * Tv + t) * 8;
#pragma unroll
      for (int j = 0; j < 8; ++j) acc += yt[j] * w[256 + j];
      sYt[r * 8 + o] = acc;
    }
    __syncthreads();

    // --- Phase F: gates = h @ W_hh^T + y_tilde @ W_ih^T + bias
    //     (M=16, N=1024, K=256): 64 n-tiles over 32 waves, 2 each.
    {
      int r = lane & 15;
      int half8 = (lane & 16) ? 8 : 0;
#pragma unroll 1
      for (int qq = 0; qq < 2; ++qq) {
        int nt = wave * 2 + qq;
        v8f acc = {};
#pragma unroll
        for (int kt = 0; kt < 8; ++kt) {
          union { v16bf v; unsigned u[8]; } a;
          int kb = kt * 32 + half8;
#pragma unroll
          for (int vv = 0; vv < 8; ++vv) {
            int k = kb + ((vv < 4) ? 2 * vv : 16 + 2 * (vv - 4));
            a.u[vv] = sHbf32[(r * 256 + k) >> 1];
          }
          union { v16bf v; U4 q[2]; } bm;
          const U4* bp = (const U4*)(packWhh + ((kt * 64 + nt) * 512 + lane * 16));
          bm.q[0] = bp[0]; bm.q[1] = bp[1];
          acc = __builtin_amdgcn_wmma_f32_16x16x32_bf16(false, a.v, false, bm.v, (short)0, acc, false, false);
        }
        int n = nt * 16 + (lane & 15);
        float bias = sBias[n];
        const float* wih = W_ih + n * 8;
#pragma unroll
        for (int vv = 0; vv < 8; ++vv) {
          int m = half8 + vv;
          const float* yv = sYt + m * 8;
          float g = acc[vv] + bias;
#pragma unroll
          for (int j = 0; j < 8; ++j) g += yv[j] * wih[j];
          sGates[m * 1024 + n] = g;
        }
      }
    }
    __syncthreads();

    // --- Phase G: LSTM pointwise update; all waves
    for (int i = tid; i < 4096; i += 1024) {
      int r = i >> 8, d = i & 255;
      float ig = sGates[r * 1024 + d];
      float fg = sGates[r * 1024 + 256 + d];
      float gg = sGates[r * 1024 + 512 + d];
      float og = sGates[r * 1024 + 768 + d];
      float cn = fast_sigmoid(fg) * sC[i] + fast_sigmoid(ig) * fast_tanh(gg);
      float hn = fast_sigmoid(og) * fast_tanh(cn);
      sC[i] = cn; sH[i] = hn;
      sCbf[i] = (__bf16)cn; sHbf[i] = (__bf16)hn;
    }
    __syncthreads();
  }

  // --- Final: out = [h, ctx] @ fcout_W^T + fcout_b
  if (tid < 128) {
    int r = tid >> 3, o = tid & 7;
    const float* w = fcout_W + o * (Dv + Ev);
    float acc = fcout_b[o];
    for (int d = 0; d < 256; ++d) acc += sH[r * 256 + d] * w[d];
    for (int e = 0; e < 256; ++e) acc += sPreCtx[r * 256 + e] * w[256 + e];
    out[(b0 + r) * 8 + o] = acc;
  }
}

// ------------------------------------------------------------- launcher ---
extern "C" void kernel_launch(void* const* d_in, const int* in_sizes, int n_in,
                              void* d_out, int out_size, void* d_ws, size_t ws_size,
                              hipStream_t stream) {
  const float* input_encoded = (const float*)d_in[0];
  const float* y_history     = (const float*)d_in[1];
  const float* h0            = (const float*)d_in[2];
  const float* c0            = (const float*)d_in[3];
  const float* attn_W1       = (const float*)d_in[4];
  const float* attn_b1       = (const float*)d_in[5];
  const float* attn_W2       = (const float*)d_in[6];
  const float* attn_b2       = (const float*)d_in[7];
  const float* W_ih          = (const float*)d_in[8];
  const float* W_hh          = (const float*)d_in[9];
  const float* b_ih          = (const float*)d_in[10];
  const float* b_hh          = (const float*)d_in[11];
  const float* fc_W          = (const float*)d_in[12];
  const float* fc_b          = (const float*)d_in[13];
  const float* fcout_W       = (const float*)d_in[14];
  const float* fcout_b       = (const float*)d_in[15];
  float* out = (float*)d_out;

  char* ws = (char*)d_ws;
  __bf16* Xbf       = (__bf16*)ws; ws += (size_t)Bv * Tv * Ev * 2;   // 16.8 MB
  __bf16* encb      = (__bf16*)ws; ws += (size_t)Bv * Tv * Ev * 2;   // 16.8 MB
  __bf16* packW1enc = (__bf16*)ws; ws += (size_t)256 * 256 * 2;
  __bf16* packW1hc  = (__bf16*)ws; ws += (size_t)512 * 256 * 2;
  __bf16* packWhh   = (__bf16*)ws; ws += (size_t)256 * 1024 * 2;

  k_f32_to_bf16<<<4096, 256, 0, stream>>>(input_encoded, Xbf, Bv * Tv * Ev);
  k_pack_b<<<64, 256, 0, stream>>>(attn_W1, packW1enc, 256, 256, 768, 512);   // W1_enc^T
  k_pack_b<<<128, 256, 0, stream>>>(attn_W1, packW1hc, 512, 256, 768, 0);     // W1_hc^T
  k_pack_b<<<256, 256, 0, stream>>>(W_hh, packWhh, 256, 1024, 256, 0);        // W_hh^T
  k_encproj<<<(Bv * Tv) / 16, 256, 0, stream>>>(Xbf, packW1enc, attn_b1, encb);
  k_scan<<<Bv / 16, 1024, 0, stream>>>(Xbf, encb, packW1hc, packWhh, y_history,
                                       h0, c0, attn_W2, attn_b2, W_ih, b_ih, b_hh,
                                       fc_W, fc_b, fcout_W, fcout_b, out);
}
